// CrossAttentionLayer_66855460930244
// MI455X (gfx1250) — compile-verified, matrix operand order checked by reading, not run on hardware
//
#include <hip/hip_runtime.h>
#include <hip/hip_bf16.h>
#include <stdint.h>

#define DMODEL 256
#define FFDIM  1024
#define NHEAD  4
#define DHEAD  64
#define NREP   128

typedef __attribute__((ext_vector_type(16))) __bf16       v16bf;
typedef __attribute__((ext_vector_type(8)))  float        v8f;
typedef __attribute__((ext_vector_type(4)))  float        f32x4;
typedef __attribute__((ext_vector_type(4)))  unsigned int u32x4;

union FragBF { u32x4 q[2]; v16bf v; };

__device__ __forceinline__ unsigned short f2bf(float f) {
  unsigned int u = __builtin_bit_cast(unsigned int, f);
  unsigned int r = (u + 0x7fffu + ((u >> 16) & 1u)) >> 16;   // RNE
  return (unsigned short)r;
}

__device__ __forceinline__ float wred_sum(float v) {
#pragma unroll
  for (int m = 16; m; m >>= 1) v += __shfl_xor(v, m, 32);
  return v;
}

// async global -> LDS copy (16B per lane), tracked by ASYNCcnt
__device__ __forceinline__ void async_ld16(unsigned lds_off, const void* g) {
  asm volatile("global_load_async_to_lds_b128 %0, %1, off"
               :: "v"(lds_off), "v"(g) : "memory");
}

// ---------------- segment boundaries (batch_index sorted) ----------------
__global__ void seg_kernel(const int* __restrict__ bi, int* s0, int* s1, int n) {
  int i = blockIdx.x * 256 + threadIdx.x;
  if (i >= n) return;
  int b = bi[i];
  if (i == 0 || bi[i - 1] != b) s0[b] = i;
  if (i == n - 1 || bi[i + 1] != b) s1[b] = i + 1;
}

// ---------------- pack fp32 weight [K x Nc] into WMMA bf16 B-fragments ----
// ktMajor==0: frag = nt*(K/32)+kt   (W1: chunk fc -> frags [16fc,16fc+16) )
// ktMajor==1: frag = kt*(Nc/16)+nt  (W2: chunk fc -> frags [16fc,16fc+16) )
// lane L holds col n = nt*16 + (L&15), half = L>>4
// elem i<8 : k = kt*32 + half*8 + i ; elem i>=8 : k = kt*32 + 16 + half*8 + (i-8)
__global__ void pack_b_kernel(const float* __restrict__ W, unsigned short* __restrict__ out,
                              int K, int Nc, int ktMajor) {
  long idx = (long)blockIdx.x * 256 + threadIdx.x;
  if (idx >= (long)K * Nc) return;
  int i = idx & 15;
  long t = idx >> 4;
  int lane = t & 31;
  long frag = t >> 5;
  int KT = K / 32, NT = Nc / 16;
  int kt, nt;
  if (ktMajor) { kt = (int)(frag / NT); nt = (int)(frag % NT); }
  else         { kt = (int)(frag % KT); nt = (int)(frag / KT); }
  int n = nt * 16 + (lane & 15);
  int half = lane >> 4;
  int k = kt * 32 + ((i < 8) ? half * 8 + i : 16 + half * 8 + (i - 8));
  out[idx] = f2bf(W[(size_t)k * Nc + n]);
}

// ---------------- t init ----------------
__global__ void init_t_kernel(const float* __restrict__ temb, float* __restrict__ t0) {
  t0[blockIdx.x * DMODEL + threadIdx.x] = temb[threadIdx.x];
}

// ---------------- rowwise LayerNorm (one wave per row, D=256) -------------
__global__ void ln_kernel(const float* __restrict__ in, float* __restrict__ out, int rows) {
  int w = threadIdx.x >> 5, lane = threadIdx.x & 31;
  int row = blockIdx.x * 8 + w;
  if (row >= rows) return;
  const float* r = in + (size_t)row * DMODEL + lane * 8;
  f32x4 a0 = *(const f32x4*)r, a1 = *(const f32x4*)(r + 4);
  float v[8] = {a0.x, a0.y, a0.z, a0.w, a1.x, a1.y, a1.z, a1.w};
  float s = 0.f, ss = 0.f;
#pragma unroll
  for (int j = 0; j < 8; ++j) { s += v[j]; ss += v[j] * v[j]; }
  s = wred_sum(s); ss = wred_sum(ss);
  float mu = s * (1.f / DMODEL);
  float var = ss * (1.f / DMODEL) - mu * mu;
  float rs = rsqrtf(var + 1e-5f);
  float* o = out + (size_t)row * DMODEL + lane * 8;
#pragma unroll
  for (int j = 0; j < 8; ++j) o[j] = (v[j] - mu) * rs;
}

// ---------------- small fp32 GEMM: out = res + act(A[MxK] @ W[KxNc]) ------
__global__ void gemm_kernel(const float* __restrict__ A, const float* __restrict__ W,
                            const float* __restrict__ res, float* __restrict__ out,
                            int M, int K, int Nc, int doRelu) {
  long idx = (long)blockIdx.x * 256 + threadIdx.x;
  if (idx >= (long)M * Nc) return;
  int n = (int)(idx % Nc);
  long m = idx / Nc;
  const float* a = A + m * K;
  const float* w = W + n;
  float acc = 0.f;
  for (int k = 0; k < K; ++k) acc += a[k] * w[(size_t)k * Nc];
  if (doRelu) acc = fmaxf(acc, 0.f);
  if (res) acc += res[idx];
  out[idx] = acc;
}

// ---------------- kq[b,h,j] = sum_dh Wk[j, h*64+dh] * q[b, h*64+dh] -------
__global__ void kq_kernel(const float* __restrict__ q, const float* __restrict__ Wk,
                          float* __restrict__ kq) {
  int b = blockIdx.x, j = threadIdx.x;
#pragma unroll
  for (int h = 0; h < NHEAD; ++h) {
    float acc = 0.f;
    const float* wr = Wk + (size_t)j * DMODEL + h * DHEAD;
    const float* qr = q + (size_t)b * DMODEL + h * DHEAD;
    for (int dh = 0; dh < DHEAD; ++dh) acc += wr[dh] * qr[dh];
    kq[((size_t)b * NHEAD + h) * DMODEL + j] = acc;
  }
}

// ---------------- fused ragged attention ----------------------------------
__global__ __launch_bounds__(256) void attn_kernel(
    const float* __restrict__ xsrc, const int* __restrict__ s0v,
    const int* __restrict__ s1v, const float* __restrict__ kq,
    float* __restrict__ attnraw) {
  __shared__ float m_sh[8][NHEAD];
  __shared__ float xs_sh[NHEAD][DMODEL];
  __shared__ float ltot[NHEAD];
  int b = blockIdx.x;
  int w = threadIdx.x >> 5, lane = threadIdx.x & 31;
  int s0 = s0v[b], s1 = s1v[b];
  float kqv[NHEAD][8];
#pragma unroll
  for (int h = 0; h < NHEAD; ++h)
#pragma unroll
    for (int j = 0; j < 8; ++j)
      kqv[h][j] = kq[((size_t)b * NHEAD + h) * DMODEL + lane * 8 + j];
  float m[NHEAD], l[NHEAD], acc[NHEAD][8];
#pragma unroll
  for (int h = 0; h < NHEAD; ++h) {
    m[h] = -1e30f; l[h] = 0.f;
#pragma unroll
    for (int j = 0; j < 8; ++j) acc[h][j] = 0.f;
  }
  for (int tok = s0 + w; tok < s1; tok += 8) {
    const float* xr = xsrc + (size_t)tok * DMODEL + lane * 8;
    f32x4 a0 = *(const f32x4*)xr, a1 = *(const f32x4*)(xr + 4);
    float v[8] = {a0.x, a0.y, a0.z, a0.w, a1.x, a1.y, a1.z, a1.w};
    float s = 0.f, ss = 0.f;
#pragma unroll
    for (int j = 0; j < 8; ++j) { s += v[j]; ss += v[j] * v[j]; }
    s = wred_sum(s); ss = wred_sum(ss);
    float mu = s * (1.f / DMODEL);
    float var = ss * (1.f / DMODEL) - mu * mu;
    float rs = rsqrtf(var + 1e-5f);
    float y[8];
#pragma unroll
    for (int j = 0; j < 8; ++j) y[j] = (v[j] - mu) * rs;
#pragma unroll
    for (int h = 0; h < NHEAD; ++h) {
      float d = 0.f;
#pragma unroll
      for (int j = 0; j < 8; ++j) d += y[j] * kqv[h][j];
      d = wred_sum(d);
      float sc = d * 0.125f;                     // 1/sqrt(64)
      float nm = fmaxf(m[h], sc);
      float f = __expf(m[h] - nm);
      float e = __expf(sc - nm);
      l[h] = l[h] * f + e;
#pragma unroll
      for (int j = 0; j < 8; ++j) acc[h][j] = acc[h][j] * f + e * y[j];
      m[h] = nm;
    }
  }
  if (lane < NHEAD) m_sh[w][lane] = m[lane];
  for (int i = threadIdx.x; i < NHEAD * DMODEL; i += 256) xs_sh[i >> 8][i & 255] = 0.f;
  if (threadIdx.x < NHEAD) ltot[threadIdx.x] = 0.f;
  __syncthreads();
  float fac[NHEAD];
#pragma unroll
  for (int h = 0; h < NHEAD; ++h) {
    float g = -1e30f;
#pragma unroll
    for (int ww = 0; ww < 8; ++ww) g = fmaxf(g, m_sh[ww][h]);
    fac[h] = (l[h] > 0.f) ? __expf(m[h] - g) : 0.f;
  }
#pragma unroll
  for (int h = 0; h < NHEAD; ++h)
#pragma unroll
    for (int j = 0; j < 8; ++j)
      atomicAdd(&xs_sh[h][lane * 8 + j], fac[h] * acc[h][j]);
  if (lane == 0)
#pragma unroll
    for (int h = 0; h < NHEAD; ++h) atomicAdd(&ltot[h], l[h] * fac[h]);
  __syncthreads();
  for (int i = threadIdx.x; i < NHEAD * DMODEL; i += 256) {
    int h = i >> 8, c = i & 255;
    attnraw[((size_t)b * NHEAD + h) * DMODEL + c] = xs_sh[h][c] / ltot[h];
  }
}

// ---------------- o1[b,e] = sum_j attnraw[b, e/64, j] * Wv[j,e] ----------
__global__ void headgemm_kernel(const float* __restrict__ attn, const float* __restrict__ Wv,
                                float* __restrict__ o1) {
  int b = blockIdx.x, e = threadIdx.x;
  int h = e >> 6;
  const float* ar = attn + ((size_t)b * NHEAD + h) * DMODEL;
  float acc = 0.f;
  for (int j = 0; j < DMODEL; ++j) acc += ar[j] * Wv[(size_t)j * DMODEL + e];
  o1[(size_t)b * DMODEL + e] = acc;
}

// ---------------- big fused WMMA kernel -----------------------------------
// x2 = (x + xo[bi]) + relu(LN(x + xo[bi]) @ W1x) @ W2x
// 8 waves/block (128 tokens), bf16 WMMA, B-fragments staged per-block into
// LDS via double-buffered global_load_async_to_lds_b128 (ASYNCcnt).
__device__ __forceinline__ void stage_chunk(const char* W1b, const char* W2b,
                                            char* bufp, int fc, int tid) {
  unsigned lds = (unsigned)(size_t)bufp + tid * 16;
  const char* g1 = W1b + (size_t)fc * 16384 + tid * 16;
  const char* g2 = W2b + (size_t)fc * 16384 + tid * 16;
#pragma unroll
  for (int i = 0; i < 4; ++i) async_ld16(lds + i * 4096, g1 + i * 4096);
#pragma unroll
  for (int i = 0; i < 4; ++i) async_ld16(lds + 16384 + i * 4096, g2 + i * 4096);
}

__global__ __launch_bounds__(256) void x_ffn_wmma_kernel(
    const float* __restrict__ xin, const int* __restrict__ bi,
    const float* __restrict__ xo, const char* __restrict__ W1b,
    const char* __restrict__ W2b, float* __restrict__ x2, int Ntot) {
  __shared__ unsigned short Abf[8][16][DMODEL];         //  64 KB LN(x) bf16
  __shared__ float Hs[8][16][32];                       //  16 KB transpose scratch
  __shared__ float Outs[8][16][DMODEL];                 // 128 KB output staging
  __shared__ __align__(16) char Bbuf[2][32768];         //  64 KB B frags (dbl buf)
  const int tid = threadIdx.x;
  const int w = tid >> 5, lane = tid & 31;
  const int row16 = (blockIdx.x * 8 + w) * 16;
  const int rlo = lane & 15, half = lane >> 4;

  stage_chunk(W1b, W2b, &Bbuf[0][0], 0, tid);           // prefetch chunk 0

  // ---- prologue: residual gather-add, LN, bf16 A tile to LDS, x2pre to HBM
  for (int r = 0; r < 16; ++r) {
    int tok = row16 + r;
    if (tok < Ntot) {
      const float* xr = xin + (size_t)tok * DMODEL + lane * 8;
      const float* xor_ = xo + (size_t)bi[tok] * DMODEL + lane * 8;
      f32x4 a0 = *(const f32x4*)xr, a1 = *(const f32x4*)(xr + 4);
      f32x4 b0 = *(const f32x4*)xor_, b1 = *(const f32x4*)(xor_ + 4);
      a0 += b0; a1 += b1;
      float* x2r = x2 + (size_t)tok * DMODEL + lane * 8;
      *(f32x4*)x2r = a0; *(f32x4*)(x2r + 4) = a1;
      float v[8] = {a0.x, a0.y, a0.z, a0.w, a1.x, a1.y, a1.z, a1.w};
      float s = 0.f, ss = 0.f;
#pragma unroll
      for (int j = 0; j < 8; ++j) { s += v[j]; ss += v[j] * v[j]; }
      s = wred_sum(s); ss = wred_sum(ss);
      float mu = s * (1.f / DMODEL);
      float var = ss * (1.f / DMODEL) - mu * mu;
      float rs = rsqrtf(var + 1e-5f);
#pragma unroll
      for (int j = 0; j < 8; ++j) Abf[w][r][lane * 8 + j] = f2bf((v[j] - mu) * rs);
    } else {
#pragma unroll
      for (int j = 0; j < 8; ++j) Abf[w][r][lane * 8 + j] = 0;
    }
  }
  asm volatile("s_wait_dscnt 0x0" ::: "memory");        // wave-local A-tile ready

  v8f outacc[16];
#pragma unroll
  for (int nt = 0; nt < 16; ++nt)
#pragma unroll
    for (int r = 0; r < 8; ++r) outacc[nt][r] = 0.f;

  for (int fc = 0; fc < FFDIM / 32; ++fc) {             // 32 ff-chunks
    if (fc + 1 < FFDIM / 32) {
      stage_chunk(W1b, W2b, &Bbuf[(fc + 1) & 1][0], fc + 1, tid);
      asm volatile("s_wait_asynccnt 0x8" ::: "memory"); // chunk fc landed
    } else {
      asm volatile("s_wait_asynccnt 0x0" ::: "memory");
    }
    __syncthreads();                                    // chunk fc visible to all
    const char* cb = &Bbuf[fc & 1][0];

    v8f h0, h1;
#pragma unroll
    for (int r = 0; r < 8; ++r) { h0[r] = 0.f; h1[r] = 0.f; }
#pragma unroll
    for (int kt = 0; kt < DMODEL / 32; ++kt) {
      FragBF a;
      a.q[0] = *(const u32x4*)&Abf[w][rlo][kt * 32 + half * 8];
      a.q[1] = *(const u32x4*)&Abf[w][rlo][kt * 32 + 16 + half * 8];
      FragBF b0f, b1f;
      const char* p0 = cb + kt * 1024 + lane * 32;
      const char* p1 = cb + (8 + kt) * 1024 + lane * 32;
      b0f.q[0] = *(const u32x4*)p0;        b0f.q[1] = *(const u32x4*)(p0 + 16);
      b1f.q[0] = *(const u32x4*)p1;        b1f.q[1] = *(const u32x4*)(p1 + 16);
      h0 = __builtin_amdgcn_wmma_f32_16x16x32_bf16(false, a.v, false, b0f.v, (short)0, h0, false, false);
      h1 = __builtin_amdgcn_wmma_f32_16x16x32_bf16(false, a.v, false, b1f.v, (short)0, h1, false, false);
    }
    // relu + lane transpose through LDS (C layout -> A layout)
#pragma unroll
    for (int r = 0; r < 8; ++r) {
      Hs[w][r + 8 * half][rlo]      = fmaxf(h0[r], 0.f);
      Hs[w][r + 8 * half][16 + rlo] = fmaxf(h1[r], 0.f);
    }
    asm volatile("s_wait_dscnt 0x0" ::: "memory");
    unsigned short tmp[16];
    const float* hr = &Hs[w][rlo][0];
#pragma unroll
    for (int j = 0; j < 8; ++j) tmp[j] = f2bf(hr[half * 8 + j]);
#pragma unroll
    for (int j = 0; j < 8; ++j) tmp[8 + j] = f2bf(hr[16 + half * 8 + j]);
    FragBF a2;
    unsigned int dw[8];
#pragma unroll
    for (int j = 0; j < 8; ++j)
      dw[j] = (unsigned)tmp[2 * j] | ((unsigned)tmp[2 * j + 1] << 16);
    a2.q[0] = (u32x4){dw[0], dw[1], dw[2], dw[3]};
    a2.q[1] = (u32x4){dw[4], dw[5], dw[6], dw[7]};
#pragma unroll
    for (int nt = 0; nt < 16; ++nt) {
      FragBF bf;
      const char* p = cb + 16384 + nt * 1024 + lane * 32;
      bf.q[0] = *(const u32x4*)p; bf.q[1] = *(const u32x4*)(p + 16);
      outacc[nt] = __builtin_amdgcn_wmma_f32_16x16x32_bf16(false, a2.v, false, bf.v, (short)0, outacc[nt], false, false);
    }
    __syncthreads();                                    // done with buf fc before refill
  }

  // ---- epilogue: C layout -> row-major via LDS, residual add, store
#pragma unroll
  for (int nt = 0; nt < 16; ++nt)
#pragma unroll
    for (int r = 0; r < 8; ++r)
      Outs[w][r + 8 * half][nt * 16 + rlo] = outacc[nt][r];
  asm volatile("s_wait_dscnt 0x0" ::: "memory");
  for (int r = 0; r < 16; ++r) {
    int tok = row16 + r;
    if (tok >= Ntot) continue;
    float* x2r = x2 + (size_t)tok * DMODEL + lane * 8;
    f32x4 p0 = *(const f32x4*)x2r, p1 = *(const f32x4*)(x2r + 4);
    const float* o = &Outs[w][r][lane * 8];
    p0 += (f32x4){o[0], o[1], o[2], o[3]};
    p1 += (f32x4){o[4], o[5], o[6], o[7]};
    *(f32x4*)x2r = p0; *(f32x4*)(x2r + 4) = p1;
  }
}

// ===========================================================================
extern "C" void kernel_launch(void* const* d_in, const int* in_sizes, int n_in,
                              void* d_out, int out_size, void* d_ws, size_t ws_size,
                              hipStream_t stream) {
  (void)n_in; (void)out_size; (void)ws_size;
  const float* x    = (const float*)d_in[0];
  const int*   bi   = (const int*)d_in[1];
  const float* temb = (const float*)d_in[2];
  // params flattened: extr dict (sorted keys) then inter dict (sorted keys)
  const float* W1_e = (const float*)d_in[3];
  const float* W2_e = (const float*)d_in[4];
  const float* Wk_e = (const float*)d_in[5];
  const float* Wo_e = (const float*)d_in[6];
  const float* Wq_e = (const float*)d_in[7];
  const float* Wv_e = (const float*)d_in[8];
  const float* W1_i = (const float*)d_in[9];
  const float* W1x  = (const float*)d_in[10];
  const float* W2_i = (const float*)d_in[11];
  const float* W2x  = (const float*)d_in[12];
  const float* Wk_i = (const float*)d_in[13];
  const float* Wo_i = (const float*)d_in[14];
  const float* Wo2  = (const float*)d_in[15];
  const float* Wq_i = (const float*)d_in[16];
  const float* Wv_i = (const float*)d_in[17];
  const float* Wv2  = (const float*)d_in[18];
  const int N = in_sizes[1];
  float* out = (float*)d_out;

  char* ws = (char*)d_ws;
  size_t off = 0;
  auto alloc = [&](size_t bytes) -> void* {
    void* p = ws + off;
    off += (bytes + 255) & ~(size_t)255;
    return p;
  };
  const size_t TB = (size_t)NREP * DMODEL * 4;
  int* seg0 = (int*)alloc(NREP * 4);
  int* seg1 = (int*)alloc(NREP * 4);
  float* t0  = (float*)alloc(TB);
  float* t1  = (float*)alloc(TB);
  float* t2  = (float*)alloc(TB);
  float* t3  = (float*)alloc(TB);
  float* tln = (float*)alloc(TB);
  float* q   = (float*)alloc(TB);
  float* o1  = (float*)alloc(TB);
  float* xov = (float*)alloc(TB);
  float* h1  = (float*)alloc((size_t)NREP * FFDIM * 4);
  float* kq  = (float*)alloc((size_t)NREP * NHEAD * DMODEL * 4);
  float* araw= (float*)alloc((size_t)NREP * NHEAD * DMODEL * 4);
  unsigned short* W1p = (unsigned short*)alloc((size_t)DMODEL * FFDIM * 2);
  unsigned short* W2p = (unsigned short*)alloc((size_t)FFDIM * DMODEL * 2);
  float* x2 = (float*)alloc((size_t)N * DMODEL * 4);

  seg_kernel<<<(N + 255) / 256, 256, 0, stream>>>(bi, seg0, seg1, N);
  pack_b_kernel<<<(DMODEL * FFDIM) / 256, 256, 0, stream>>>(W1x, W1p, DMODEL, FFDIM, 0);
  pack_b_kernel<<<(FFDIM * DMODEL) / 256, 256, 0, stream>>>(W2x, W2p, FFDIM, DMODEL, 1);
  init_t_kernel<<<NREP, DMODEL, 0, stream>>>(temb, t0);

  // ---------------- inter layer ----------------
  ln_kernel<<<NREP / 8, 256, 0, stream>>>(t0, tln, NREP);
  gemm_kernel<<<(NREP * DMODEL) / 256, 256, 0, stream>>>(tln, Wq_i, nullptr, q, NREP, DMODEL, DMODEL, 0);
  kq_kernel<<<NREP, DMODEL, 0, stream>>>(q, Wk_i, kq);
  attn_kernel<<<NREP, 256, 0, stream>>>(x, seg0, seg1, kq, araw);
  headgemm_kernel<<<NREP, DMODEL, 0, stream>>>(araw, Wv_i, o1);
  gemm_kernel<<<(NREP * DMODEL) / 256, 256, 0, stream>>>(o1, Wo_i, t0, t1, NREP, DMODEL, DMODEL, 0);
  ln_kernel<<<NREP / 8, 256, 0, stream>>>(t1, tln, NREP);
  gemm_kernel<<<(NREP * FFDIM) / 256, 256, 0, stream>>>(tln, W1_i, nullptr, h1, NREP, DMODEL, FFDIM, 1);
  gemm_kernel<<<(NREP * DMODEL) / 256, 256, 0, stream>>>(h1, W2_i, t1, t2, NREP, FFDIM, DMODEL, 0);
  ln_kernel<<<NREP / 8, 256, 0, stream>>>(t2, tln, NREP);
  gemm_kernel<<<(NREP * DMODEL) / 256, 256, 0, stream>>>(tln, Wv2, nullptr, o1, NREP, DMODEL, DMODEL, 0);
  gemm_kernel<<<(NREP * DMODEL) / 256, 256, 0, stream>>>(o1, Wo2, nullptr, xov, NREP, DMODEL, DMODEL, 0);
  x_ffn_wmma_kernel<<<(N + 127) / 128, 256, 0, stream>>>(x, bi, xov, (const char*)W1p, (const char*)W2p, x2, N);

  // ---------------- extr layer ----------------
  ln_kernel<<<NREP / 8, 256, 0, stream>>>(t2, tln, NREP);
  gemm_kernel<<<(NREP * DMODEL) / 256, 256, 0, stream>>>(tln, Wq_e, nullptr, q, NREP, DMODEL, DMODEL, 0);
  kq_kernel<<<NREP, DMODEL, 0, stream>>>(q, Wk_e, kq);
  attn_kernel<<<NREP, 256, 0, stream>>>(x2, seg0, seg1, kq, araw);
  headgemm_kernel<<<NREP, DMODEL, 0, stream>>>(araw, Wv_e, o1);
  gemm_kernel<<<(NREP * DMODEL) / 256, 256, 0, stream>>>(o1, Wo_e, t2, t3, NREP, DMODEL, DMODEL, 0);
  ln_kernel<<<NREP / 8, 256, 0, stream>>>(t3, tln, NREP);
  gemm_kernel<<<(NREP * FFDIM) / 256, 256, 0, stream>>>(tln, W1_e, nullptr, h1, NREP, DMODEL, FFDIM, 1);
  gemm_kernel<<<(NREP * DMODEL) / 256, 256, 0, stream>>>(h1, W2_e, t3, out, NREP, FFDIM, DMODEL, 0);
}
